// NeighborAware_52707838656551
// MI455X (gfx1250) — matrix operand order repaired and checked by reading, not compiled, not run on hardware
//
#include <hip/hip_runtime.h>

typedef __attribute__((ext_vector_type(16))) _Float16 v16h;
typedef __attribute__((ext_vector_type(8)))  _Float16 v8h;
typedef __attribute__((ext_vector_type(4)))  _Float16 v4h;
typedef __attribute__((ext_vector_type(8)))  float    v8f;
typedef __attribute__((ext_vector_type(4)))  float    v4f;

#define EMB  128
#define KNB  5
#define SEQ  6
#define TB   32            // batch elements per tower workgroup
#define ROWS (TB * SEQ)    // 192 sequence rows per workgroup
#define MB   64            // batch elements per MLP workgroup

// ---------------------------------------------------------------------------
// CDNA5 async global->LDS copy (ASYNCcnt path) via inline asm.
// VDST = per-lane LDS byte address (low 32 bits of generic shared pointer),
// VADDR = 64-bit global address, off = null SADDR.
// ---------------------------------------------------------------------------
#if defined(__gfx1250__)
#define HAVE_ASYNC_LDS 1
#else
#define HAVE_ASYNC_LDS 0
#endif

#if HAVE_ASYNC_LDS
__device__ __forceinline__ void async_copy16(const void* g, void* s) {
  unsigned lds = (unsigned)(size_t)s;                 // LDS byte offset
  unsigned long long ga = (unsigned long long)(size_t)g;
  asm volatile("global_load_async_to_lds_b128 %0, %1, off"
               :
               : "v"(lds), "v"(ga)
               : "memory");
}
__device__ __forceinline__ void wait_async() {
  asm volatile("s_wait_asynccnt 0x0" ::: "memory");
}
#endif

// ---------------------------------------------------------------------------
// WMMA helpers (CDNA5 16x16x32 f16 -> f32)
// ---------------------------------------------------------------------------
__device__ __forceinline__ v8f wmma16(v16h a, v16h b, v8f c) {
  return __builtin_amdgcn_wmma_f32_16x16x32_f16(false, a, false, b, (short)0, c,
                                                false, false);
}

// A fragment: 16x32 f16 tile, row-major source with row stride `ldk` halves.
// Lane l: row = l&15, koff = (l&16)?16:0; K = koff+j.
__device__ __forceinline__ v16h load_a_frag(const _Float16* tileBase, int lane,
                                            int ldk) {
  int row  = lane & 15;
  int koff = (lane & 16) ? 16 : 0;
  const _Float16* p = tileBase + (size_t)row * ldk + koff;
  v8h lo = *reinterpret_cast<const v8h*>(p);        // K = koff .. koff+7
  v8h hi = *reinterpret_cast<const v8h*>(p + 8);    // K = koff+8 .. koff+15
  v16h r;
#pragma unroll
  for (int j = 0; j < 8; ++j) { r[j] = lo[j]; r[j + 8] = hi[j]; }
  return r;
}

// B fragment: 32x16 f16 from row-major W[out,in]; B[k][n] = W[nBase+n][kBase+k].
// Lane l holds column (l&15), 16 consecutive K (K half by lane half).
__device__ __forceinline__ v16h load_b_fragW(const _Float16* W, int nFeatBase,
                                             int kBase, int ldw, int lane) {
  int n  = lane & 15;
  int kq = (lane & 16) ? 16 : 0;
  const _Float16* p = W + (size_t)(nFeatBase + n) * ldw + kBase + kq;
  return *reinterpret_cast<const v16h*>(p);
}

// C/D: elem r -> M = r + (lane>=16 ? 8:0), N = lane&15.
template <bool RELU>
__device__ __forceinline__ void store_c_f16(_Float16* tileOut, int ldc,
                                            int lane, v8f c, float bias) {
  int n  = lane & 15;
  int mo = (lane & 16) ? 8 : 0;
#pragma unroll
  for (int r = 0; r < 8; ++r) {
    float v = c[r] + bias;
    if (RELU) v = fmaxf(v, 0.0f);
    tileOut[(size_t)(mo + r) * ldc + n] = (_Float16)v;
  }
}

// ---------------------------------------------------------------------------
// f32 -> f16 weight conversion
// ---------------------------------------------------------------------------
__global__ void f32_to_f16_kernel(const float* __restrict__ src,
                                  _Float16* __restrict__ dst, int n) {
  int i = blockIdx.x * 256 + threadIdx.x;
  if (i < n) dst[i] = (_Float16)src[i];
}

// ---------------------------------------------------------------------------
// Fused tower: gather -> QKV GEMM -> masked softmax attention -> out-proj.
// One workgroup handles TB=32 batch elements (192 sequence rows).
// ---------------------------------------------------------------------------
__global__ __launch_bounds__(256) void tower_kernel(
    const int* __restrict__ ids, const float* __restrict__ table,
    const int* __restrict__ topk, const _Float16* __restrict__ in_w,
    const float* __restrict__ in_b, const _Float16* __restrict__ out_w,
    const float* __restrict__ out_b, _Float16* __restrict__ hbuf, int colOff) {
  __shared__ __align__(16) _Float16 sA[ROWS * EMB];   // 48 KB (reused for ctx)
  __shared__ __align__(16) _Float16 sKV[ROWS * 256];  // 96 KB (also f32 stage)
  __shared__ __align__(16) float    sQ[TB * EMB];     // 16 KB
  __shared__ float sMask[ROWS];
  __shared__ int   sRow[ROWS];

  const int tid   = threadIdx.x;
  const int lane  = tid & 31;
  const int wave  = tid >> 5;
  const int tile0 = blockIdx.x * TB;

  // --- row ids + attention mask -------------------------------------------
  for (int r = tid; r < ROWS; r += 256) {
    int e = r / SEQ, t = r % SEQ;
    int id = ids[tile0 + e];
    int rowid;
    float m = 0.0f;
    if (t == 0) {
      rowid = id;
    } else {
      rowid = topk[(size_t)id * KNB + (t - 1)];
      if (rowid == 0) m = -1e30f;
    }
    sRow[r]  = rowid;
    sMask[r] = m;
  }
  __syncthreads();

  // --- gather embedding rows, f32 -> f16 into sA --------------------------
#if HAVE_ASYNC_LDS
  {
    float* sStage = reinterpret_cast<float*>(sKV);  // 192*128 f32 == 96 KB
    for (int idx = tid; idx < ROWS * 32; idx += 256) {
      int r = idx >> 5, c = idx & 31;  // 16-byte chunk c of row r
      async_copy16(table + (size_t)sRow[r] * EMB + c * 4,
                   sStage + (size_t)r * EMB + c * 4);
    }
    wait_async();
    __syncthreads();
    for (int idx = tid; idx < ROWS * 32; idx += 256) {
      int r = idx >> 5, c4 = (idx & 31) << 2;
      v4f v = *reinterpret_cast<const v4f*>(sStage + (size_t)r * EMB + c4);
      v4h h;
#pragma unroll
      for (int j = 0; j < 4; ++j) h[j] = (_Float16)v[j];
      *reinterpret_cast<v4h*>(sA + (size_t)r * EMB + c4) = h;
    }
  }
#else
  for (int idx = tid; idx < ROWS * 32; idx += 256) {
    int r = idx >> 5, c4 = (idx & 31) << 2;
    v4f v = *reinterpret_cast<const v4f*>(table + (size_t)sRow[r] * EMB + c4);
    v4h h;
#pragma unroll
    for (int j = 0; j < 4; ++j) h[j] = (_Float16)v[j];
    *reinterpret_cast<v4h*>(sA + (size_t)r * EMB + c4) = h;
  }
#endif
  __syncthreads();

  // --- K,V GEMM: [192 x 128] x [128 x 256]; B fragments held in VGPRs -----
  for (int nt = wave; nt < 16; nt += 8) {
    v16h b[4];
#pragma unroll
    for (int kt = 0; kt < 4; ++kt)
      b[kt] = load_b_fragW(in_w, EMB + nt * 16, kt * 32, EMB, lane);
    float bias = in_b[EMB + nt * 16 + (lane & 15)];
    for (int mt = 0; mt < ROWS / 16; ++mt) {
      v8f c = {};
#pragma unroll
      for (int kt = 0; kt < 4; ++kt)
        c = wmma16(load_a_frag(sA + (size_t)mt * 16 * EMB + kt * 32, lane, EMB),
                   b[kt], c);
      store_c_f16<false>(sKV + (size_t)mt * 16 * 256 + nt * 16, 256, lane, c,
                         bias);
    }
  }

  // --- Q GEMM (token-0 rows only): [32 x 128] x [128 x 128] ---------------
  {
    const int nt = wave;  // 8 waves == 8 N tiles
    v16h b[4];
#pragma unroll
    for (int kt = 0; kt < 4; ++kt)
      b[kt] = load_b_fragW(in_w, nt * 16, kt * 32, EMB, lane);
    float bias = in_b[nt * 16 + (lane & 15)];
#pragma unroll
    for (int mt = 0; mt < 2; ++mt) {
      v8f c = {};
#pragma unroll
      for (int kt = 0; kt < 4; ++kt)
        c = wmma16(load_a_frag(sA + (size_t)mt * 16 * SEQ * EMB + kt * 32, lane,
                               SEQ * EMB),
                   b[kt], c);
      int n = lane & 15, mo = (lane & 16) ? 8 : 0;
#pragma unroll
      for (int r = 0; r < 8; ++r)
        sQ[(size_t)(mt * 16 + mo + r) * EMB + nt * 16 + n] = c[r] + bias;
    }
  }
  __syncthreads();

  // --- attention: 8 lanes per element, 16 dims per lane -------------------
  {
    const int e = tid >> 3;
    const int dbase = (tid & 7) << 4;
    const float invs = 0.08838834764831845f;  // 1/sqrt(128)
    float qr[16];
#pragma unroll
    for (int d = 0; d < 16; ++d) qr[d] = sQ[(size_t)e * EMB + dbase + d];

    float sc[SEQ];
#pragma unroll
    for (int t = 0; t < SEQ; ++t) {
      const _Float16* krow = sKV + (size_t)(e * SEQ + t) * 256;
      float acc = 0.0f;
#pragma unroll
      for (int d = 0; d < 16; ++d) acc += qr[d] * (float)krow[dbase + d];
      acc += __shfl_xor(acc, 1, 8);
      acc += __shfl_xor(acc, 2, 8);
      acc += __shfl_xor(acc, 4, 8);
      sc[t] = acc * invs + sMask[e * SEQ + t];
    }
    float mx = sc[0];
#pragma unroll
    for (int t = 1; t < SEQ; ++t) mx = fmaxf(mx, sc[t]);
    float w[SEQ], sum = 0.0f;
#pragma unroll
    for (int t = 0; t < SEQ; ++t) { w[t] = __expf(sc[t] - mx); sum += w[t]; }
    float inv = 1.0f / sum;
    // ctx written into the (now free) sA region: ctx[e][d] = sA[e*128+d]
#pragma unroll
    for (int d = 0; d < 16; ++d) {
      float acc = 0.0f;
#pragma unroll
      for (int t = 0; t < SEQ; ++t)
        acc += w[t] * (float)sKV[(size_t)(e * SEQ + t) * 256 + EMB + dbase + d];
      sA[(size_t)e * EMB + dbase + d] = (_Float16)(acc * inv);
    }
  }
  __syncthreads();

  // --- out-proj: [32 x 128] x [128 x 128] -> h[B,256] (half) --------------
  {
    const int nt = wave;
    v16h b[4];
#pragma unroll
    for (int kt = 0; kt < 4; ++kt)
      b[kt] = load_b_fragW(out_w, nt * 16, kt * 32, EMB, lane);
    float bias = out_b[nt * 16 + (lane & 15)];
#pragma unroll
    for (int mt = 0; mt < 2; ++mt) {
      v8f c = {};
#pragma unroll
      for (int kt = 0; kt < 4; ++kt)
        c = wmma16(load_a_frag(sA + (size_t)mt * 16 * EMB + kt * 32, lane, EMB),
                   b[kt], c);
      int n = lane & 15, mo = (lane & 16) ? 8 : 0;
#pragma unroll
      for (int r = 0; r < 8; ++r) {
        size_t row = (size_t)(tile0 + mt * 16 + mo + r);
        hbuf[row * 256 + colOff + nt * 16 + n] = (_Float16)(c[r] + bias);
      }
    }
  }
}

// ---------------------------------------------------------------------------
// Fused MLP: h[B,256] -> relu(W1) -> relu(W2) -> W3 dot -> out[B]
// ---------------------------------------------------------------------------
__global__ __launch_bounds__(256) void mlp_kernel(
    const _Float16* __restrict__ hbuf, const _Float16* __restrict__ W1,
    const float* __restrict__ b1, const _Float16* __restrict__ W2,
    const float* __restrict__ b2, const float* __restrict__ W3,
    const float* __restrict__ b3, float* __restrict__ out) {
  __shared__ __align__(16) _Float16 sH[MB * 256];   // 32 KB
  __shared__ __align__(16) _Float16 sH1[MB * 128];  // 16 KB
  __shared__ __align__(16) _Float16 sH2[MB * 64];   //  8 KB

  const int tid  = threadIdx.x;
  const int lane = tid & 31;
  const int wave = tid >> 5;
  const size_t base = (size_t)blockIdx.x * MB;

  // load h tile (2048 x 16B chunks)
#if HAVE_ASYNC_LDS
  for (int i = tid; i < MB * 256 / 8; i += 256)
    async_copy16(hbuf + base * 256 + (size_t)i * 8, sH + (size_t)i * 8);
  wait_async();
#else
  {
    const v8h* src = reinterpret_cast<const v8h*>(hbuf + base * 256);
    v8h* dst = reinterpret_cast<v8h*>(sH);
    for (int i = tid; i < MB * 256 / 8; i += 256) dst[i] = src[i];
  }
#endif
  __syncthreads();

  // layer 1: [64 x 256] x [256 x 128], relu;  nt = wave
  {
    const int nt = wave;
    v16h b[8];
#pragma unroll
    for (int kt = 0; kt < 8; ++kt)
      b[kt] = load_b_fragW(W1, nt * 16, kt * 32, 256, lane);
    float bias = b1[nt * 16 + (lane & 15)];
#pragma unroll
    for (int mt = 0; mt < 4; ++mt) {
      v8f c = {};
#pragma unroll
      for (int kt = 0; kt < 8; ++kt)
        c = wmma16(load_a_frag(sH + (size_t)mt * 16 * 256 + kt * 32, lane, 256),
                   b[kt], c);
      store_c_f16<true>(sH1 + (size_t)mt * 16 * 128 + nt * 16, 128, lane, c,
                        bias);
    }
  }
  __syncthreads();

  // layer 2: [64 x 128] x [128 x 64], relu; nt = wave&3, mt in {w>>2, w>>2+2}
  {
    const int nt  = wave & 3;
    const int mt0 = wave >> 2;
    v16h b[4];
#pragma unroll
    for (int kt = 0; kt < 4; ++kt)
      b[kt] = load_b_fragW(W2, nt * 16, kt * 32, 128, lane);
    float bias = b2[nt * 16 + (lane & 15)];
#pragma unroll
    for (int mm = 0; mm < 2; ++mm) {
      int mt = mt0 + mm * 2;
      v8f c = {};
#pragma unroll
      for (int kt = 0; kt < 4; ++kt)
        c = wmma16(
            load_a_frag(sH1 + (size_t)mt * 16 * 128 + kt * 32, lane, 128),
            b[kt], c);
      store_c_f16<true>(sH2 + (size_t)mt * 16 * 64 + nt * 16, 64, lane, c,
                        bias);
    }
  }
  __syncthreads();

  // layer 3: dot(h2[e], W3) + b3 ; 4 lanes per element
  {
    int e = tid >> 2;
    int dbase = (tid & 3) << 4;
    float acc = 0.0f;
#pragma unroll
    for (int d = 0; d < 16; ++d)
      acc += (float)sH2[(size_t)e * 64 + dbase + d] * W3[dbase + d];
    acc += __shfl_xor(acc, 1, 4);
    acc += __shfl_xor(acc, 2, 4);
    if ((tid & 3) == 0) out[base + e] = acc + b3[0];
  }
}

// ---------------------------------------------------------------------------
extern "C" void kernel_launch(void* const* d_in, const int* in_sizes, int n_in,
                              void* d_out, int out_size, void* d_ws,
                              size_t ws_size, hipStream_t stream) {
  (void)n_in; (void)out_size; (void)ws_size;
  const int*   user       = (const int*)d_in[0];
  const int*   item       = (const int*)d_in[1];
  const float* user_table = (const float*)d_in[2];
  const float* item_table = (const float*)d_in[3];
  const int*   user_topk  = (const int*)d_in[4];
  const int*   item_topk  = (const int*)d_in[5];
  const float* u_in_w  = (const float*)d_in[6];
  const float* u_in_b  = (const float*)d_in[7];
  const float* u_out_w = (const float*)d_in[8];
  const float* u_out_b = (const float*)d_in[9];
  const float* i_in_w  = (const float*)d_in[10];
  const float* i_in_b  = (const float*)d_in[11];
  const float* i_out_w = (const float*)d_in[12];
  const float* i_out_b = (const float*)d_in[13];
  const float* W1 = (const float*)d_in[14];
  const float* b1 = (const float*)d_in[15];
  const float* W2 = (const float*)d_in[16];
  const float* b2 = (const float*)d_in[17];
  const float* W3 = (const float*)d_in[18];
  const float* b3 = (const float*)d_in[19];
  float* out = (float*)d_out;
  const int B = in_sizes[0];

  _Float16* ws      = (_Float16*)d_ws;
  _Float16* u_in_h  = ws;
  _Float16* i_in_h  = u_in_h + 3 * EMB * EMB;
  _Float16* u_out_h = i_in_h + 3 * EMB * EMB;
  _Float16* i_out_h = u_out_h + EMB * EMB;
  _Float16* W1h     = i_out_h + EMB * EMB;
  _Float16* W2h     = W1h + 128 * 256;
  _Float16* hbuf    = W2h + 64 * 128;  // B * 256 halves

  auto cvt = [&](const float* s, _Float16* d, int n) {
    f32_to_f16_kernel<<<dim3((n + 255) / 256), dim3(256), 0, stream>>>(s, d, n);
  };
  cvt(u_in_w, u_in_h, 3 * EMB * EMB);
  cvt(i_in_w, i_in_h, 3 * EMB * EMB);
  cvt(u_out_w, u_out_h, EMB * EMB);
  cvt(i_out_w, i_out_h, EMB * EMB);
  cvt(W1, W1h, 128 * 256);
  cvt(W2, W2h, 64 * 128);

  tower_kernel<<<dim3(B / TB), dim3(256), 0, stream>>>(
      user, user_table, user_topk, u_in_h, u_in_b, u_out_h, u_out_b, hbuf, 0);
  tower_kernel<<<dim3(B / TB), dim3(256), 0, stream>>>(
      item, item_table, item_topk, i_in_h, i_in_b, i_out_h, i_out_b, hbuf, EMB);
  mlp_kernel<<<dim3(B / MB), dim3(256), 0, stream>>>(hbuf, W1h, b1, W2h, b2, W3,
                                                     b3, out);
}
